// Decoder_61813169324317
// MI455X (gfx1250) — matrix operand order repaired and checked by reading, not compiled, not run on hardware
//
#include <hip/hip_runtime.h>

#define LEAK 0.2f

typedef __attribute__((ext_vector_type(16))) __bf16 v16bf;
typedef __attribute__((ext_vector_type(8)))  __bf16 v8bf;
typedef __attribute__((ext_vector_type(8)))  float  v8f;

__device__ __forceinline__ float leaky_(float x) { return x >= 0.f ? x : LEAK * x; }
__device__ __forceinline__ float sigmoid_(float x) { return 1.f / (1.f + expf(-x)); }

__device__ __forceinline__ unsigned short f2bf(float f) {
    unsigned u = __float_as_uint(f);
    unsigned r = u + 0x7FFFu + ((u >> 16) & 1u);
    return (unsigned short)(r >> 16);
}

// ---------------------------------------------------------------- init
__global__ void init_kernel(unsigned* cnt, const float* __restrict__ hidden, float* __restrict__ hb) {
    int i = blockIdx.x * blockDim.x + threadIdx.x;
    if (i < 3) cnt[i] = 0u;
    if (i < 3072) hb[i] = hidden[i];
}

__global__ void copy_kernel(const float* __restrict__ x, float* __restrict__ y, int n) {
    int i = blockIdx.x * blockDim.x + threadIdx.x;
    if (i < n) y[i] = x[i];
}

// ---------------------------------------------------------------- attention
__global__ void qkv_kernel(const float* __restrict__ qsrc, int qdim,
                           const float* __restrict__ kvsrc, int kvdim,
                           const float* __restrict__ Wq, const float* __restrict__ bq,
                           const float* __restrict__ Wk, const float* __restrict__ bk,
                           const float* __restrict__ Wv, const float* __restrict__ bv,
                           float* __restrict__ Q, float* __restrict__ K, float* __restrict__ V) {
    int t = blockIdx.x * blockDim.x + threadIdx.x;
    if (t >= 1024) return;
    float qin[8], kin[8];
    for (int j = 0; j < qdim; ++j) qin[j] = qsrc[t * qdim + j];
    for (int j = 0; j < kvdim; ++j) kin[j] = kvsrc[t * kvdim + j];
    for (int e = 0; e < 4; ++e) {
        float sq = bq[e], sk = bk[e], sv = bv[e];
        for (int j = 0; j < qdim; ++j) sq += Wq[e * qdim + j] * qin[j];
        for (int j = 0; j < kvdim; ++j) {
            sk += Wk[e * kvdim + j] * kin[j];
            sv += Wv[e * kvdim + j] * kin[j];
        }
        Q[t * 4 + e] = sq; K[t * 4 + e] = sk; V[t * 4 + e] = sv;
    }
}

// online-softmax attention over S=1024, H_ATT=2 heads of Dh=2, fused
// output-proj + residual + leaky + LayerNorm(4)
__global__ void __launch_bounds__(256) attn_kernel(
    const float* __restrict__ Q, const float* __restrict__ K, const float* __restrict__ V,
    const float* __restrict__ resid,
    const float* __restrict__ Wo, const float* __restrict__ bo,
    const float* __restrict__ g, const float* __restrict__ bln,
    float* __restrict__ outp) {
    __shared__ float Ks[4096];
    __shared__ float Vs[4096];
    for (int i = threadIdx.x; i < 4096; i += 256) { Ks[i] = K[i]; Vs[i] = V[i]; }
    __syncthreads();
    int t = blockIdx.x * 256 + threadIdx.x;
    float o[4];
    for (int h = 0; h < 2; ++h) {
        float q0 = Q[t * 4 + h * 2], q1 = Q[t * 4 + h * 2 + 1];
        float m = -1e30f, l = 0.f, a0 = 0.f, a1 = 0.f;
        for (int s = 0; s < 1024; ++s) {
            float sc = (q0 * Ks[s * 4 + h * 2] + q1 * Ks[s * 4 + h * 2 + 1]) * 0.70710678f;
            float mn = fmaxf(m, sc);
            float corr = expf(m - mn);
            float p = expf(sc - mn);
            l = l * corr + p;
            a0 = a0 * corr + p * Vs[s * 4 + h * 2];
            a1 = a1 * corr + p * Vs[s * 4 + h * 2 + 1];
            m = mn;
        }
        o[h * 2] = a0 / l; o[h * 2 + 1] = a1 / l;
    }
    float y[4];
    for (int e = 0; e < 4; ++e) {
        float s = bo[e];
        for (int j = 0; j < 4; ++j) s += Wo[e * 4 + j] * o[j];
        y[e] = leaky_(s + resid[t * 4 + e]);
    }
    float mu = 0.25f * (y[0] + y[1] + y[2] + y[3]);
    float var = 0.f;
    for (int e = 0; e < 4; ++e) { float d = y[e] - mu; var += d * d; }
    var *= 0.25f;
    float inv = rsqrtf(var + 1e-5f);
    for (int e = 0; e < 4; ++e)
        outp[t * 4 + e] = (y[e] - mu) * inv * g[e] + bln[e];
}

__global__ void ffn_kernel(const float* __restrict__ n2,
                           const float* __restrict__ Wl1, const float* __restrict__ bl1,
                           const float* __restrict__ Wl2, const float* __restrict__ bl2,
                           const float* __restrict__ g, const float* __restrict__ bln,
                           float* __restrict__ outp) {
    int t = blockIdx.x * blockDim.x + threadIdx.x;
    if (t >= 1024) return;
    float xin[4], h1[4], h2[4], y[4];
    for (int e = 0; e < 4; ++e) xin[e] = n2[t * 4 + e];
    for (int e = 0; e < 4; ++e) {
        float s = bl1[e];
        for (int j = 0; j < 4; ++j) s += Wl1[e * 4 + j] * xin[j];
        h1[e] = leaky_(s);
    }
    for (int e = 0; e < 4; ++e) {
        float s = bl2[e];
        for (int j = 0; j < 4; ++j) s += Wl2[e * 4 + j] * h1[j];
        h2[e] = leaky_(s);
    }
    for (int e = 0; e < 4; ++e) y[e] = leaky_(h2[e] + xin[e]);
    float mu = 0.25f * (y[0] + y[1] + y[2] + y[3]);
    float var = 0.f;
    for (int e = 0; e < 4; ++e) { float d = y[e] - mu; var += d * d; }
    var *= 0.25f;
    float inv = rsqrtf(var + 1e-5f);
    for (int e = 0; e < 4; ++e)
        outp[t * 4 + e] = (y[e] - mu) * inv * g[e] + bln[e];
}

// ---------------------------------------------------------------- CNN
__global__ void transpose_enc(const float* __restrict__ enc, float* __restrict__ ctx) {
    int i = blockIdx.x * blockDim.x + threadIdx.x;
    if (i >= 6144) return;
    int c = i / 1024, l = i % 1024;
    ctx[c * 1024 + l] = enc[l * 6 + c];
}

__global__ void conv1d_leaky(const float* __restrict__ x, int C, int L,
                             const float* __restrict__ W, const float* __restrict__ b,
                             int O, int K, float* __restrict__ y) {
    int Lo = L - K + 1;
    int idx = blockIdx.x * blockDim.x + threadIdx.x;
    if (idx >= O * Lo) return;
    int o = idx / Lo, l = idx % Lo;
    float s = b[o];
    for (int i = 0; i < C; ++i)
        for (int k = 0; k < K; ++k)
            s += W[(o * C + i) * K + k] * x[i * L + l + k];
    y[o * Lo + l] = leaky_(s);
}

__global__ void maxpool2_kernel(const float* __restrict__ x, int C, int L, float* __restrict__ y) {
    int Lo = L / 2;
    int idx = blockIdx.x * blockDim.x + threadIdx.x;
    if (idx >= C * Lo) return;
    int c = idx / Lo, l = idx % Lo;
    y[c * Lo + l] = fmaxf(x[c * L + 2 * l], x[c * L + 2 * l + 1]);
}

// ---------------------------------------------------------------- GRU
// layer-0 input projection (K=4): gi[t][j] = bih[j] + sum_k x[t][k]*Wih[j][k]
__global__ void gi_small_kernel(const float* __restrict__ x, const float* __restrict__ Wih,
                                const float* __restrict__ bih, float* __restrict__ gi) {
    int idx = blockIdx.x * blockDim.x + threadIdx.x;
    if (idx >= 1024 * 3072) return;
    int t = idx / 3072, j = idx % 3072;
    float s = bih[j];
    for (int k = 0; k < 4; ++k) s += x[t * 4 + k] * Wih[j * 4 + k];
    gi[(size_t)t * 3072 + j] = s;
}

__global__ void f32_to_bf16_kernel(const float* __restrict__ x, unsigned short* __restrict__ y, int n) {
    int i = blockIdx.x * blockDim.x + threadIdx.x;
    if (i < n) y[i] = f2bf(x[i]);
}

// C(M x N) = A(M x K bf16, row-major) * Wt(N x K bf16, row-major)^T + bias
// Block covers 16 (M) x 512 (N); each wave owns 4 16x16 C tiles so the A
// fragment is reused 4x per K step. The 16 x K A tile is DMA-staged into LDS
// once per block via GLOBAL_LOAD_ASYNC_TO_LDS_B128 (ASYNCcnt), rows padded to
// 2064 B (16 distinct banks for the 16 row readers). The K loop is manually
// double-buffered: A/B fragments for step k+32 are prefetched into registers
// while the 4 WMMAs for step k issue.
__global__ void __launch_bounds__(256) gemm_bf16_wmma(
    const unsigned short* __restrict__ Au, const unsigned short* __restrict__ Wu,
    const float* __restrict__ bias, float* __restrict__ C,
    int M, int N, int Kd) {
    constexpr int ROWB = 1032;                      // padded row stride (ushorts)
    __shared__ __align__(16) unsigned short Atile[16 * ROWB];  // ~33 KB
    const __bf16* Wt = (const __bf16*)Wu;
    int wave = threadIdx.x >> 5;
    int lane = threadIdx.x & 31;
    int mtiles = M / 16;
    int bm = blockIdx.x % mtiles;
    int bn = blockIdx.x / mtiles;
    int tileM = bm * 16;
    int baseN = bn * 512 + wave * 16;
    int l15 = lane & 15;
    int hi = lane >> 4;  // A: 0->K 0..15, 1->K 16..31; B: 0->K 0..15, 1->K 16..31

    // ---- async-stage A tile (16 rows x Kd) into LDS, 16B chunks ----
    {
        int cpr = Kd / 8;                    // 16B chunks per row
        int chunks = 16 * cpr;
        for (int idx = threadIdx.x; idx < chunks; idx += 256) {
            int row = idx / cpr;
            int col = idx % cpr;
            const unsigned short* gsrc = Au + (size_t)(tileM + row) * Kd + col * 8;
            unsigned ldst = (unsigned)(uintptr_t)&Atile[row * ROWB + col * 8];
            asm volatile("global_load_async_to_lds_b128 %0, %1, off"
                         :: "v"(ldst), "v"(gsrc)
                         : "memory");
        }
        asm volatile("s_wait_asynccnt 0x0" ::: "memory");
    }
    __syncthreads();

    const unsigned short* As = &Atile[l15 * ROWB];
    const __bf16* brow[4];
#pragma unroll
    for (int j = 0; j < 4; ++j)
        brow[j] = Wt + (size_t)(baseN + j * 128 + l15) * Kd;

    auto load_a = [&](int k) -> v16bf {
        v8bf alo = *(const v8bf*)((const __bf16*)As + k + hi * 16);
        v8bf ahi = *(const v8bf*)((const __bf16*)As + k + hi * 16 + 8);
        v16bf a;
#pragma unroll
        for (int i = 0; i < 8; ++i) { a[i] = alo[i]; a[8 + i] = ahi[i]; }
        return a;
    };
    auto load_b = [&](int j, int k) -> v16bf {
        return *(const v16bf*)(brow[j] + k + hi * 16);
    };

    v8f c[4];
#pragma unroll
    for (int j = 0; j < 4; ++j)
#pragma unroll
        for (int i = 0; i < 8; ++i) c[j][i] = 0.f;

    v16bf a_cur = load_a(0);
    v16bf b0 = load_b(0, 0), b1 = load_b(1, 0), b2 = load_b(2, 0), b3 = load_b(3, 0);
    for (int k = 0; k < Kd - 32; k += 32) {
        int kn = k + 32;
        v16bf a_nxt = load_a(kn);
        v16bf n0 = load_b(0, kn), n1 = load_b(1, kn), n2 = load_b(2, kn), n3 = load_b(3, kn);
        c[0] = __builtin_amdgcn_wmma_f32_16x16x32_bf16(false, a_cur, false, b0, (short)0, c[0], false, false);
        c[1] = __builtin_amdgcn_wmma_f32_16x16x32_bf16(false, a_cur, false, b1, (short)0, c[1], false, false);
        c[2] = __builtin_amdgcn_wmma_f32_16x16x32_bf16(false, a_cur, false, b2, (short)0, c[2], false, false);
        c[3] = __builtin_amdgcn_wmma_f32_16x16x32_bf16(false, a_cur, false, b3, (short)0, c[3], false, false);
        a_cur = a_nxt; b0 = n0; b1 = n1; b2 = n2; b3 = n3;
    }
    c[0] = __builtin_amdgcn_wmma_f32_16x16x32_bf16(false, a_cur, false, b0, (short)0, c[0], false, false);
    c[1] = __builtin_amdgcn_wmma_f32_16x16x32_bf16(false, a_cur, false, b1, (short)0, c[1], false, false);
    c[2] = __builtin_amdgcn_wmma_f32_16x16x32_bf16(false, a_cur, false, b2, (short)0, c[2], false, false);
    c[3] = __builtin_amdgcn_wmma_f32_16x16x32_bf16(false, a_cur, false, b3, (short)0, c[3], false, false);

#pragma unroll
    for (int j = 0; j < 4; ++j) {
        int col = baseN + j * 128 + l15;
        float bv = bias[col];
#pragma unroll
        for (int r = 0; r < 8; ++r) {
            int row = tileM + r + hi * 8;
            C[(size_t)row * N + col] = c[j][r] + bv;
        }
    }
}

__device__ __forceinline__ void grid_sync(unsigned* cnt, unsigned target) {
    __threadfence();
    __syncthreads();
    if (threadIdx.x == 0) {
        __hip_atomic_fetch_add(cnt, 1u, __ATOMIC_ACQ_REL, __HIP_MEMORY_SCOPE_AGENT);
        while (__hip_atomic_load(cnt, __ATOMIC_ACQUIRE, __HIP_MEMORY_SCOPE_AGENT) < target)
            __builtin_amdgcn_s_sleep(8);
    }
    __syncthreads();
}

// Persistent GRU scan. Grid = 64 WGs x 256 threads; WG owns 16 hidden units.
// Per step: 48 dot-1024 products (3 gates x 16), each by a 16-lane group with
// coalesced weight reads; Whh stays L2-resident across all 1024 steps.
__global__ void __launch_bounds__(256) gru_scan_kernel(
    const float* __restrict__ Whh, const float* __restrict__ bhh,
    const float* __restrict__ gi, float* __restrict__ h,
    float* __restrict__ out, unsigned* cnt) {
    const int T = 1024, H = 1024;
    int base = blockIdx.x * 16;
    int grp = threadIdx.x >> 4;
    int lane = threadIdx.x & 15;
    __shared__ float gh[3][16];
    __shared__ float hs[1024];
    for (int t = 0; t < T; ++t) {
        for (int i = threadIdx.x; i < H; i += 256) hs[i] = h[i];
        __syncthreads();
        for (int g = 0; g < 3; ++g) {
            int i = base + grp;
            int row = g * H + i;
            const float* w = Whh + (size_t)row * H;
            float s = 0.f;
            for (int k = lane; k < H; k += 16) s += w[k] * hs[k];
            for (int off = 8; off; off >>= 1) s += __shfl_xor(s, off, 16);
            if (lane == 0) gh[g][grp] = s + bhh[row];
        }
        __syncthreads();
        if (threadIdx.x < 16) {
            int i = base + threadIdx.x;
            const float* git = gi + (size_t)t * 3 * H;
            float r = sigmoid_(git[i] + gh[0][threadIdx.x]);
            float z = sigmoid_(git[H + i] + gh[1][threadIdx.x]);
            float n = tanhf(git[2 * H + i] + r * gh[2][threadIdx.x]);
            float hn = (1.f - z) * n + z * hs[i];
            h[i] = hn;
            out[(size_t)t * H + i] = hn;
        }
        grid_sync(cnt, (unsigned)gridDim.x * (unsigned)(t + 1));
    }
}

// ---------------------------------------------------------------- MLP head
// y[row] = act( dot(Arows[row], x) + bias[row*bias_stride] )
__global__ void __launch_bounds__(256) matvec_rows(
    const float* __restrict__ Arows, int K, const float* __restrict__ x,
    const float* __restrict__ bias, int bias_stride,
    float* __restrict__ y, int apply_leaky) {
    __shared__ float red[256];
    int row = blockIdx.x;
    const float* a = Arows + (size_t)row * K;
    float s = 0.f;
    for (int k = threadIdx.x; k < K; k += 256) s += a[k] * x[k];
    red[threadIdx.x] = s;
    __syncthreads();
    for (int off = 128; off; off >>= 1) {
        if (threadIdx.x < off) red[threadIdx.x] += red[threadIdx.x + off];
        __syncthreads();
    }
    if (threadIdx.x == 0) {
        float v = red[0] + bias[row * bias_stride];
        y[row] = apply_leaky ? leaky_(v) : v;
    }
}

__global__ void __launch_bounds__(256) add_ln_kernel(
    const float* __restrict__ a, const float* __restrict__ b,
    const float* __restrict__ g, const float* __restrict__ bb, float* __restrict__ y) {
    __shared__ float s1[256], s2[256];
    float lsum = 0.f, lsq = 0.f;
    for (int i = threadIdx.x; i < 1024; i += 256) {
        float v = a[i] + b[i];
        lsum += v; lsq += v * v;
    }
    s1[threadIdx.x] = lsum; s2[threadIdx.x] = lsq;
    __syncthreads();
    for (int off = 128; off; off >>= 1) {
        if (threadIdx.x < off) { s1[threadIdx.x] += s1[threadIdx.x + off]; s2[threadIdx.x] += s2[threadIdx.x + off]; }
        __syncthreads();
    }
    float mu = s1[0] * (1.f / 1024.f);
    float var = s2[0] * (1.f / 1024.f) - mu * mu;
    float inv = rsqrtf(var + 1e-5f);
    for (int i = threadIdx.x; i < 1024; i += 256)
        y[i] = (a[i] + b[i] - mu) * inv * g[i] + bb[i];
}

// ---------------------------------------------------------------- launch
static inline int cdiv(int a, int b) { return (a + b - 1) / b; }

extern "C" void kernel_launch(void* const* d_in, const int* in_sizes, int n_in,
                              void* d_out, int out_size, void* d_ws, size_t ws_size,
                              hipStream_t stream) {
    auto F = [&](int i) { return (const float*)d_in[i]; };

    // ---- workspace layout
    char* w = (char*)d_ws;
    size_t off = 0;
    auto alloc = [&](size_t bytes) -> void* {
        off = (off + 255) & ~(size_t)255;
        void* p = w + off;
        off += bytes;
        return p;
    };
    unsigned* cnt = (unsigned*)alloc(3 * sizeof(unsigned));
    float* Q = (float*)alloc(4096 * 4);
    float* K = (float*)alloc(4096 * 4);
    float* V = (float*)alloc(4096 * 4);
    float* n1b = (float*)alloc(4096 * 4);
    float* n2b = (float*)alloc(4096 * 4);
    float* attA = (float*)alloc(4096 * 4);
    float* attB = (float*)alloc(4096 * 4);
    float* cbA = (float*)alloc(16384 * 4);
    float* cbB = (float*)alloc(16384 * 4);
    float* gi = (float*)alloc((size_t)1024 * 3072 * 4);
    float* out0 = (float*)alloc((size_t)1024 * 1024 * 4);
    float* out1 = (float*)alloc((size_t)1024 * 1024 * 4);
    float* out2 = (float*)alloc((size_t)1024 * 1024 * 4);
    float* hb = (float*)alloc(3 * 1024 * 4);
    unsigned short* Abf = (unsigned short*)alloc((size_t)1024 * 1024 * 2);
    unsigned short* Wb1 = (unsigned short*)alloc((size_t)3072 * 1024 * 2);
    unsigned short* Wb2 = (unsigned short*)alloc((size_t)3072 * 1024 * 2);
    float* xcnn = (float*)alloc(1024 * 4);
    float* xatt = (float*)alloc(1024 * 4);
    float* xln = (float*)alloc(1024 * 4);
    float* h3 = (float*)alloc(128 * 4);
    float* h4 = (float*)alloc(64 * 4);

    const float* enc = F(1);
    const float* dec = F(2);

    init_kernel<<<cdiv(3072, 256), 256, 0, stream>>>(cnt, F(0), hb);

    // ---- attention stack (3 blocks)
    const float* cur = dec;
    float* outs[3] = { attA, attB, attA };
    for (int p = 0; p < 3; ++p) {
        int bi = 3 + 26 * p;
        // self-attention
        qkv_kernel<<<4, 256, 0, stream>>>(cur, 4, cur, 4,
            F(bi + 0), F(bi + 1), F(bi + 2), F(bi + 3), F(bi + 4), F(bi + 5), Q, K, V);
        attn_kernel<<<4, 256, 0, stream>>>(Q, K, V, cur,
            F(bi + 6), F(bi + 7), F(bi + 8), F(bi + 9), n1b);
        // cross-attention (k,v from enc, dim 6)
        qkv_kernel<<<4, 256, 0, stream>>>(n1b, 4, enc, 6,
            F(bi + 10), F(bi + 11), F(bi + 12), F(bi + 13), F(bi + 14), F(bi + 15), Q, K, V);
        attn_kernel<<<4, 256, 0, stream>>>(Q, K, V, n1b,
            F(bi + 16), F(bi + 17), F(bi + 18), F(bi + 19), n2b);
        // FFN + LN
        ffn_kernel<<<4, 256, 0, stream>>>(n2b,
            F(bi + 20), F(bi + 21), F(bi + 22), F(bi + 23), F(bi + 24), F(bi + 25), outs[p]);
        cur = outs[p];
    }
    const float* attFinal = cur;  // attA

    // ---- CNN branch
    transpose_enc<<<cdiv(6144, 256), 256, 0, stream>>>(enc, cbA);
    conv1d_leaky<<<cdiv(6 * 1014, 256), 256, 0, stream>>>(cbA, 6, 1024, F(81), F(82), 6, 11, cbB);
    conv1d_leaky<<<cdiv(12 * 1004, 256), 256, 0, stream>>>(cbB, 6, 1014, F(83), F(84), 12, 11, cbA);
    maxpool2_kernel<<<cdiv(12 * 502, 256), 256, 0, stream>>>(cbA, 12, 1004, cbB);
    conv1d_leaky<<<cdiv(12 * 498, 256), 256, 0, stream>>>(cbB, 12, 502, F(85), F(86), 12, 5, cbA);
    conv1d_leaky<<<cdiv(24 * 494, 256), 256, 0, stream>>>(cbA, 12, 498, F(87), F(88), 24, 5, cbB);
    maxpool2_kernel<<<cdiv(24 * 247, 256), 256, 0, stream>>>(cbB, 24, 494, cbA);
    conv1d_leaky<<<cdiv(24 * 245, 256), 256, 0, stream>>>(cbA, 24, 247, F(89), F(90), 24, 3, cbB);
    conv1d_leaky<<<cdiv(24 * 243, 256), 256, 0, stream>>>(cbB, 24, 245, F(91), F(92), 24, 3, cbA);
    maxpool2_kernel<<<cdiv(24 * 121, 256), 256, 0, stream>>>(cbA, 24, 243, cbB);
    const float* ctxflat = cbB;  // 2904 floats

    // ---- GRU (3 layers): precompute input projections, then persistent scan
    // layer 0 (K=4, VALU)
    gi_small_kernel<<<cdiv(1024 * 3072, 256), 256, 0, stream>>>(attFinal, F(93), F(95), gi);
    gru_scan_kernel<<<64, 256, 0, stream>>>(F(94), F(96), gi, hb + 0, out0, cnt + 0);

    // layer 1: gi1 = out0 @ Wih1^T + bih1 via bf16 WMMA
    f32_to_bf16_kernel<<<cdiv(1024 * 1024, 256), 256, 0, stream>>>(out0, Abf, 1024 * 1024);
    f32_to_bf16_kernel<<<cdiv(3072 * 1024, 256), 256, 0, stream>>>(F(97), Wb1, 3072 * 1024);
    gemm_bf16_wmma<<<(1024 / 16) * (3072 / 512), 256, 0, stream>>>(Abf, Wb1, F(99), gi, 1024, 3072, 1024);
    gru_scan_kernel<<<64, 256, 0, stream>>>(F(98), F(100), gi, hb + 1024, out1, cnt + 1);

    // layer 2
    f32_to_bf16_kernel<<<cdiv(1024 * 1024, 256), 256, 0, stream>>>(out1, Abf, 1024 * 1024);
    f32_to_bf16_kernel<<<cdiv(3072 * 1024, 256), 256, 0, stream>>>(F(101), Wb2, 3072 * 1024);
    gemm_bf16_wmma<<<(1024 / 16) * (3072 / 512), 256, 0, stream>>>(Abf, Wb2, F(103), gi, 1024, 3072, 1024);
    gru_scan_kernel<<<64, 256, 0, stream>>>(F(102), F(104), gi, hb + 2048, out2, cnt + 2);

    // ---- MLP head
    matvec_rows<<<1024, 256, 0, stream>>>(F(105), 2904, ctxflat, F(106), 1, xcnn, 1);   // W1a
    matvec_rows<<<1024, 256, 0, stream>>>(out2, 1024, F(107), F(108), 0, xatt, 1);      // W1b
    add_ln_kernel<<<1, 256, 0, stream>>>(xatt, xcnn, F(109), F(110), xln);
    matvec_rows<<<100, 256, 0, stream>>>(F(111), 1024, xln, F(112), 1, h3, 1);          // W3
    matvec_rows<<<50, 256, 0, stream>>>(F(113), 100, h3, F(114), 1, h4, 1);             // W4
    matvec_rows<<<4, 256, 0, stream>>>(F(115), 50, h4, F(116), 1, (float*)d_out, 0);    // W5 -> x(4)

    // gru_hidden -> d_out[4 .. 4+3072)
    copy_kernel<<<cdiv(3072, 256), 256, 0, stream>>>(hb, (float*)d_out + 4, 3072);
}
